// EncFeaturesClassifier_79697413144910
// MI455X (gfx1250) — matrix operand order repaired and checked by reading, not compiled, not run on hardware
//
#include <hip/hip_runtime.h>
#include <cstdint>
#include <cstddef>

// ---------------------------------------------------------------------------
// Types
// ---------------------------------------------------------------------------
typedef __bf16 bf16_t;
typedef __attribute__((ext_vector_type(16))) __bf16 v16bf;
typedef __attribute__((ext_vector_type(8)))  __bf16 v8bf;
typedef __attribute__((ext_vector_type(4)))  __bf16 v4bf;
typedef __attribute__((ext_vector_type(8)))  float  v8f;

#define Bdim 512
#define Sdim 128
#define Hdim 512
#define TVdim 1024
#define GH3  1536   // 3*H

__device__ __forceinline__ bf16_t f2bf(float f) {
  union { float f; uint32_t u; } c; c.f = f;
  uint32_t r = c.u + 0x7FFFu + ((c.u >> 16) & 1u);   // round-to-nearest-even
  union { unsigned short s; bf16_t b; } o; o.s = (unsigned short)(r >> 16);
  return o.b;
}
__device__ __forceinline__ float bf2f(bf16_t b) {
  union { unsigned short s; bf16_t b; } i; i.b = b;
  union { uint32_t u; float f; } o; o.u = ((uint32_t)i.s) << 16;
  return o.f;
}
__device__ __forceinline__ float sigmoidf_(float x) { return 1.0f / (1.0f + __expf(-x)); }

// ---------------------------------------------------------------------------
// WMMA fragment helpers (CDNA5 16x16x32 bf16, wave32)
// A: 16x32 row-major tile. lane<16: M=lane, K 0..7 & 16..23; lane>=16: M=lane-16, K 8..15 & 24..31
// ---------------------------------------------------------------------------
__device__ __forceinline__ v16bf load_a_frag(const bf16_t* Abase, int lda) {
  int lane = threadIdx.x & 31;
  const bf16_t* p = Abase + (lane & 15) * lda + ((lane >> 4) << 3);
  v8bf lo = *(const v8bf*)p;
  v8bf hi = *(const v8bf*)(p + 16);
  return __builtin_shufflevector(lo, hi, 0,1,2,3,4,5,6,7,8,9,10,11,12,13,14,15);
}
// B from W [N x K] row-major (computes A @ W^T). lane = N column; lane<16: K 0..7 & 16..23, lane>=16: K 8..15 & 24..31
__device__ __forceinline__ v16bf load_b_frag(const bf16_t* Wtile, int ldw) {
  int lane = threadIdx.x & 31;
  const bf16_t* p = Wtile + (lane & 15) * ldw + ((lane >> 4) << 3);
  v8bf lo = *(const v8bf*)p;
  v8bf hi = *(const v8bf*)(p + 16);
  return __builtin_shufflevector(lo, hi, 0,1,2,3,4,5,6,7,8,9,10,11,12,13,14,15);
}
__device__ __forceinline__ v8f wmma_bf16(v16bf a, v16bf b, v8f c) {
  return __builtin_amdgcn_wmma_f32_16x16x32_bf16(false, a, false, b, (short)0, c, false, false);
}

// ---------------------------------------------------------------------------
// fp32 -> bf16 conversion
// ---------------------------------------------------------------------------
__global__ void cvt_f32_bf16_kernel(const float* __restrict__ src,
                                    bf16_t* __restrict__ dst, int n) {
  for (int i = blockIdx.x * blockDim.x + threadIdx.x; i < n; i += gridDim.x * blockDim.x)
    dst[i] = f2bf(src[i]);
}

// ---------------------------------------------------------------------------
// feats = concat(vec_features, wemb0[w0], wemb1[w1], wemb2[w2]) -> bf16 [B,512]
// ---------------------------------------------------------------------------
__global__ void build_feats_kernel(const float* __restrict__ vec,
                                   const int* __restrict__ wf,
                                   const float* __restrict__ we0,
                                   const float* __restrict__ we1,
                                   const float* __restrict__ we2,
                                   bf16_t* __restrict__ feats) {
  int b = blockIdx.x;
  int j = threadIdx.x;           // 0..127
  bf16_t* row = feats + b * 512;
  row[j]       = f2bf(vec[b * 128 + j]);
  row[128 + j] = f2bf(we0[(size_t)wf[b * 3 + 0] * 128 + j]);
  row[256 + j] = f2bf(we1[(size_t)wf[b * 3 + 1] * 128 + j]);
  row[384 + j] = f2bf(we2[(size_t)wf[b * 3 + 2] * 128 + j]);
}

// ---------------------------------------------------------------------------
// x_bf[b*S+s, :] = bf16(relu(goal_emb[goals[b,s]]))    (grid = B*S, 128 thr)
// ---------------------------------------------------------------------------
__global__ void gather_x_kernel(const int* __restrict__ goals,
                                const float* __restrict__ goal_emb,
                                bf16_t* __restrict__ xbf) {
  int row = blockIdx.x;                       // 0 .. B*S-1  (== b*S + s)
  int g = goals[row];
  const float* src = goal_emb + (size_t)g * Hdim;
  bf16_t* dst = xbf + (size_t)row * Hdim;
  int j = threadIdx.x * 4;                    // 128 threads x 4 cols = 512
  float4 v = *(const float4*)(src + j);
  v4bf o;
  o[0] = f2bf(v.x > 0.0f ? v.x : 0.0f);
  o[1] = f2bf(v.y > 0.0f ? v.y : 0.0f);
  o[2] = f2bf(v.z > 0.0f ? v.z : 0.0f);
  o[3] = f2bf(v.w > 0.0f ? v.w : 0.0f);
  *(v4bf*)(dst + j) = o;
}

// ---------------------------------------------------------------------------
// Generic WMMA GEMM: out = act(A[MxK]bf16 @ W[NxK]^T + bias)
// grid = (N/128, M/16), block = 128 (4 waves, two 16x16 N-tiles per wave).
// Full 16xK A-strip staged to LDS once -> barrier-free K loop, 2 WMMA per
// A-fragment load.
// ---------------------------------------------------------------------------
__global__ __launch_bounds__(128)
void wmma_gemm_kernel(const bf16_t* __restrict__ A, int lda,
                      const bf16_t* __restrict__ W, int ldw,
                      const float* __restrict__ bias,
                      float* __restrict__ outF, bf16_t* __restrict__ outB,
                      int ldo, int ocol, int relu, int K) {
  __shared__ bf16_t sA[16 * 1024];            // 32 KB max (K <= 1024)
  int wave = threadIdx.x >> 5;
  int lane = threadIdx.x & 31;
  int m0 = blockIdx.y * 16;
  int nbase = blockIdx.x * 128 + wave * 32;

  // stage full 16xK A strip (16B per thread per iter, no inner-loop barriers)
  {
    int r = threadIdx.x >> 3;                 // 16 rows, 8 threads per row
    for (int c = (threadIdx.x & 7) * 8; c < K; c += 64)
      *(v8bf*)(sA + r * K + c) = *(const v8bf*)(A + (size_t)(m0 + r) * lda + c);
  }
  __syncthreads();

  const bf16_t* W0 = W + (size_t)nbase * ldw;
  const bf16_t* W1 = W + (size_t)(nbase + 16) * ldw;
  v8f acc[2] = {{}, {}};
  for (int k0 = 0; k0 < K; k0 += 32) {
    __builtin_prefetch((const void*)(W0 + k0 + 128), 0, 1);
    v16bf a = load_a_frag(sA + k0, K);
    acc[0] = wmma_bf16(a, load_b_frag(W0 + k0, ldw), acc[0]);
    acc[1] = wmma_bf16(a, load_b_frag(W1 + k0, ldw), acc[1]);
  }

  int n = lane & 15;
  int moff = (lane >> 4) * 8;
#pragma unroll
  for (int tile = 0; tile < 2; ++tile) {
    int nn = nbase + tile * 16 + n;
    float bv = bias ? bias[nn] : 0.0f;
#pragma unroll
    for (int r = 0; r < 8; ++r) {
      float v = acc[tile][r] + bv;
      if (relu) v = v > 0.0f ? v : 0.0f;
      size_t idx = (size_t)(m0 + moff + r) * ldo + ocol + nn;
      if (outF) outF[idx] = v;
      if (outB) outB[idx] = f2bf(v);
    }
  }
}

// ---------------------------------------------------------------------------
// GRU recurrence (hidden-side only; gi precomputed). grid = B/16 blocks,
// 1024 threads (32 waves, one hidden 16-col tile each). h lives in LDS bf16.
// Per step: 3 gate chains of WMMA over K=512, then fused gate math using
// gi streamed from global (prefetched one step ahead).
// Writes relu(h_final) into full[:, 0:512].
// ---------------------------------------------------------------------------
__global__ __launch_bounds__(1024)
void gru_rec_kernel(const bf16_t* __restrict__ gi,    // [B, S, 1536] bf16
                    const bf16_t* __restrict__ Whh,   // [1536, 512] bf16
                    const float* __restrict__ bih,    // [1536]
                    const float* __restrict__ bhh,    // [1536]
                    bf16_t* __restrict__ full_bf) {   // [B, 1024]
  __shared__ bf16_t sH[16 * Hdim];                    // 16 KB
  int wave = threadIdx.x >> 5;    // 0..31 == hidden tile
  int lane = threadIdx.x & 31;
  int b0 = blockIdx.x * 16;
  int n = lane & 15;
  int moff = (lane >> 4) * 8;
  int col = wave * 16 + n;        // hidden unit 0..511

  for (int i = threadIdx.x; i < 16 * Hdim; i += blockDim.x) sH[i] = f2bf(0.0f);

  float f_bir = bih[col],        f_bhr = bhh[col];
  float f_biz = bih[512 + col],  f_bhz = bhh[512 + col];
  float f_bin = bih[1024 + col], f_bhn = bhh[1024 + col];

  const bf16_t* wr = Whh + (size_t)(wave * 16) * Hdim;
  const bf16_t* wz = Whh + (size_t)(512 + wave * 16) * Hdim;
  const bf16_t* wn = Whh + (size_t)(1024 + wave * 16) * Hdim;

  __syncthreads();

  for (int t = 0; t < Sdim; ++t) {
    if (t + 1 < Sdim)
      __builtin_prefetch((const void*)(gi + ((size_t)(b0 + moff) * Sdim + t + 1) * GH3 + col), 0, 1);

    v8f ar = {}, az = {}, an = {};
    for (int k0 = 0; k0 < Hdim; k0 += 32) {
      v16bf ah = load_a_frag(sH + k0, Hdim);
      ar = wmma_bf16(ah, load_b_frag(wr + k0, Hdim), ar);
      az = wmma_bf16(ah, load_b_frag(wz + k0, Hdim), az);
      an = wmma_bf16(ah, load_b_frag(wn + k0, Hdim), an);
    }

    float newh[8];
#pragma unroll
    for (int r = 0; r < 8; ++r) {
      const bf16_t* gir = gi + ((size_t)(b0 + moff + r) * Sdim + t) * GH3;
      float g_r = bf2f(gir[col]);
      float g_z = bf2f(gir[512 + col]);
      float g_n = bf2f(gir[1024 + col]);
      float rg = sigmoidf_(g_r + f_bir + ar[r] + f_bhr);
      float zg = sigmoidf_(g_z + f_biz + az[r] + f_bhz);
      float ng = tanhf(g_n + f_bin + rg * (an[r] + f_bhn));
      float hp = bf2f(sH[(moff + r) * Hdim + col]);
      newh[r] = (1.0f - zg) * ng + zg * hp;
    }
    __syncthreads();            // all waves done reading sH
#pragma unroll
    for (int r = 0; r < 8; ++r)
      sH[(moff + r) * Hdim + col] = f2bf(newh[r]);
    __syncthreads();            // writes visible before next step's WMMA
  }

  // full[:, 0:512] = relu(h_final)
  for (int i = threadIdx.x; i < 16 * Hdim; i += blockDim.x) {
    int r = i >> 9;
    int c = i & 511;
    float v = bf2f(sH[i]);
    full_bf[(b0 + r) * TVdim + c] = f2bf(v > 0.0f ? v : 0.0f);
  }
}

// ---------------------------------------------------------------------------
// Row-wise log_softmax over [B, 1024]
// ---------------------------------------------------------------------------
__global__ __launch_bounds__(256)
void log_softmax_kernel(const float* __restrict__ logits, float* __restrict__ out) {
  __shared__ float red[256];
  int b = blockIdx.x;
  int tid = threadIdx.x;
  const float* row = logits + (size_t)b * TVdim;
  float mx = -3.4e38f;
  for (int i = tid; i < TVdim; i += 256) mx = fmaxf(mx, row[i]);
  red[tid] = mx; __syncthreads();
  for (int s = 128; s > 0; s >>= 1) {
    if (tid < s) red[tid] = fmaxf(red[tid], red[tid + s]);
    __syncthreads();
  }
  mx = red[0]; __syncthreads();
  float sum = 0.0f;
  for (int i = tid; i < TVdim; i += 256) sum += __expf(row[i] - mx);
  red[tid] = sum; __syncthreads();
  for (int s = 128; s > 0; s >>= 1) {
    if (tid < s) red[tid] += red[tid + s];
    __syncthreads();
  }
  float lse = __logf(red[0]) + mx;
  for (int i = tid; i < TVdim; i += 256) out[(size_t)b * TVdim + i] = row[i] - lse;
}

// ---------------------------------------------------------------------------
// Host launch
// ---------------------------------------------------------------------------
static void launch_gemm(const bf16_t* A, int lda, const bf16_t* W, int ldw,
                        const float* bias, float* outF, bf16_t* outB,
                        int ldo, int ocol, int relu, int M, int N, int K,
                        hipStream_t s) {
  dim3 grid(N / 128, M / 16);
  wmma_gemm_kernel<<<grid, 128, 0, s>>>(A, lda, W, ldw, bias, outF, outB, ldo, ocol, relu, K);
}

static void launch_cvt(const float* src, bf16_t* dst, int n, hipStream_t s) {
  int blocks = (n + 255) / 256;
  if (blocks > 2048) blocks = 2048;
  cvt_f32_bf16_kernel<<<blocks, 256, 0, s>>>(src, dst, n);
}

extern "C" void kernel_launch(void* const* d_in, const int* in_sizes, int n_in,
                              void* d_out, int out_size, void* d_ws, size_t ws_size,
                              hipStream_t stream) {
  const float* vec       = (const float*)d_in[0];
  const int*   wf        = (const int*)d_in[1];
  const int*   goals     = (const int*)d_in[2];
  const float* wemb0     = (const float*)d_in[3];
  const float* wemb1     = (const float*)d_in[4];
  const float* wemb2     = (const float*)d_in[5];
  const float* feat_in_W = (const float*)d_in[6];
  const float* feat_in_b = (const float*)d_in[7];
  const float* enc_W     = (const float*)d_in[8];
  const float* enc_b     = (const float*)d_in[9];
  const float* goal_emb  = (const float*)d_in[10];
  const float* gru_Wih   = (const float*)d_in[11];
  const float* gru_Whh   = (const float*)d_in[12];
  const float* gru_bih   = (const float*)d_in[13];
  const float* gru_bhh   = (const float*)d_in[14];
  const float* dec_in_W  = (const float*)d_in[15];
  const float* dec_in_b  = (const float*)d_in[16];
  const float* dec_out_W = (const float*)d_in[17];
  const float* dec_out_b = (const float*)d_in[18];
  float* out = (float*)d_out;

  char* ws = (char*)d_ws;
  size_t off = 0;
  auto wsalloc = [&](size_t bytes) -> void* {
    void* p = ws + off;
    off = (off + bytes + 255) & ~(size_t)255;
    return p;
  };

  bf16_t* wfeat_bf   = (bf16_t*)wsalloc((size_t)512 * 512 * 2);
  bf16_t* wenc_bf    = (bf16_t*)wsalloc((size_t)2 * 512 * 512 * 2);
  bf16_t* wdecin_bf  = (bf16_t*)wsalloc((size_t)512 * 1024 * 2);
  bf16_t* wdecout_bf = (bf16_t*)wsalloc((size_t)1024 * 512 * 2);
  bf16_t* wih_bf     = (bf16_t*)wsalloc((size_t)GH3 * 512 * 2);
  bf16_t* whh_bf     = (bf16_t*)wsalloc((size_t)GH3 * 512 * 2);
  bf16_t* feats_bf   = (bf16_t*)wsalloc((size_t)Bdim * 512 * 2);
  bf16_t* buf0_bf    = (bf16_t*)wsalloc((size_t)Bdim * 512 * 2);
  bf16_t* buf1_bf    = (bf16_t*)wsalloc((size_t)Bdim * 512 * 2);
  bf16_t* full_bf    = (bf16_t*)wsalloc((size_t)Bdim * 1024 * 2);
  bf16_t* hid2_bf    = (bf16_t*)wsalloc((size_t)Bdim * 512 * 2);
  float*  logits_f   = (float*)wsalloc((size_t)Bdim * 1024 * 4);
  bf16_t* x_bf       = (bf16_t*)wsalloc((size_t)Bdim * Sdim * Hdim * 2);   // 64 MB
  bf16_t* gi_bf      = (bf16_t*)wsalloc((size_t)Bdim * Sdim * GH3 * 2);    // 201 MB

  // 1) convert weights fp32 -> bf16
  launch_cvt(feat_in_W, wfeat_bf,   512 * 512, stream);
  launch_cvt(enc_W,     wenc_bf,    2 * 512 * 512, stream);
  launch_cvt(dec_in_W,  wdecin_bf,  512 * 1024, stream);
  launch_cvt(dec_out_W, wdecout_bf, 1024 * 512, stream);
  launch_cvt(gru_Wih,   wih_bf,     GH3 * 512, stream);
  launch_cvt(gru_Whh,   whh_bf,     GH3 * 512, stream);

  // 2) x = relu(goal_emb[goals]) -> bf16 [B*S, 512]
  gather_x_kernel<<<Bdim * Sdim, 128, 0, stream>>>(goals, goal_emb, x_bf);

  // 3) gi = x @ Wih^T : fully parallel 65536x1536x512 WMMA GEMM (fills chip;
  //    removes half the FLOPs from the 128-step serial recurrence path)
  launch_gemm(x_bf, 512, wih_bf, 512, nullptr, nullptr, gi_bf,
              GH3, 0, 0, Bdim * Sdim, GH3, 512, stream);

  // 4) feature concat + MLP (WMMA); last layer writes relu into full[:,512:]
  build_feats_kernel<<<Bdim, 128, 0, stream>>>(vec, wf, wemb0, wemb1, wemb2, feats_bf);
  launch_gemm(feats_bf, 512, wfeat_bf, 512, feat_in_b, nullptr, buf0_bf,
              512, 0, 1, Bdim, 512, 512, stream);
  launch_gemm(buf0_bf, 512, wenc_bf, 512, enc_b, nullptr, buf1_bf,
              512, 0, 1, Bdim, 512, 512, stream);
  launch_gemm(buf1_bf, 512, wenc_bf + (size_t)512 * 512, 512, enc_b + 512,
              nullptr, full_bf, 1024, 512, 1, Bdim, 512, 512, stream);

  // 5) GRU recurrence (hidden side only); writes relu(h) into full[:,0:512]
  gru_rec_kernel<<<Bdim / 16, 1024, 0, stream>>>(gi_bf, whh_bf, gru_bih, gru_bhh, full_bf);

  // 6) decoder: relu(full @ dec_in^T + b) -> logits = hid2 @ dec_out^T + b
  launch_gemm(full_bf, 1024, wdecin_bf, 1024, dec_in_b, nullptr, hid2_bf,
              512, 0, 1, Bdim, 512, 1024, stream);
  launch_gemm(hid2_bf, 512, wdecout_bf, 512, dec_out_b, logits_f, nullptr,
              1024, 0, 0, Bdim, 1024, 512, stream);

  // 7) log_softmax rows -> d_out
  log_softmax_kernel<<<Bdim, 256, 0, stream>>>(logits_f, out);
}
